// GCNModel_3332894622178
// MI455X (gfx1250) — matrix operand order repaired and checked by reading, not compiled, not run on hardware
//
#include <hip/hip_runtime.h>
#include <cstdint>
#include <cstddef>

// ---------------------------------------------------------------------------
// CDNA5 (gfx1250) GCN pipeline: bf16 WMMA GEMMs + atomic edge aggregation.
// ---------------------------------------------------------------------------

typedef __attribute__((ext_vector_type(16))) __bf16 v16bf;
typedef __attribute__((ext_vector_type(8)))  float  v8f;

#define HPAD 64          // padded feature stride (floats) for hidden buffers
#define NTILES 4         // HPAD / 16
#define FIN 512
#define F1 50
#define F2 40

__device__ __forceinline__ unsigned int f2bf(float f) {
  unsigned int u = __float_as_uint(f);
  u += 0x7FFFu + ((u >> 16) & 1u);   // round-to-nearest-even to bf16
  return u >> 16;
}

union ABvec { v16bf v; unsigned int u[8]; };

// ---------------------------------------------------------------------------
// Pack a (Klog x Nlog) row-major fp32 weight matrix into the per-lane WMMA
// B-operand layout (bf16, 16x16x32).  Index:  ((nt*ktiles + kt)*32 + lane)*8
// dwords; lane<16 holds K=kt*32+{0..15}, lane>=16 holds K=kt*32+{16..31},
// VGPR j packs pair (2j, 2j+1) low/high.
// ---------------------------------------------------------------------------
__global__ void pack_w_kernel(const float* __restrict__ W, int Klog, int Nlog,
                              int ktiles, unsigned int* __restrict__ out) {
  int t = blockIdx.x * blockDim.x + threadIdx.x;
  int total = NTILES * ktiles * 32;
  if (t >= total) return;
  int lane = t & 31;
  int kt   = (t >> 5) % ktiles;
  int nt   = t / (32 * ktiles);
  int n    = nt * 16 + (lane & 15);
  int kb   = kt * 32 + ((lane >> 4) << 4);   // 0 or +16
  unsigned int* o = out + (size_t)t * 8;
#pragma unroll
  for (int j = 0; j < 8; ++j) {
    int k0 = kb + 2 * j, k1 = k0 + 1;
    float f0 = (k0 < Klog && n < Nlog) ? W[(size_t)k0 * Nlog + n] : 0.0f;
    float f1 = (k1 < Klog && n < Nlog) ? W[(size_t)k1 * Nlog + n] : 0.0f;
    o[j] = f2bf(f0) | (f2bf(f1) << 16);
  }
}

// ---------------------------------------------------------------------------
// WMMA GEMM:  C[M x HPAD] = A[M x lda(fp32)] * Bpacked   (bf16 in, f32 acc)
// One wave computes a 16x64 tile; K looped in steps of 32.  All four B
// operands are fetched before the WMMA burst so loads clause together and
// the four v_wmma issue back-to-back behind a single wait.
// ---------------------------------------------------------------------------
__global__ void __launch_bounds__(256)
wmma_gemm_kernel(const float* __restrict__ A, int lda, int ktiles,
                 const unsigned int* __restrict__ Bp,
                 float* __restrict__ C, int M) {
  const int lane = threadIdx.x & 31;
  const int wave = blockIdx.x * (blockDim.x >> 5) + (threadIdx.x >> 5);
  const int m0 = wave << 4;
  if (m0 >= M) return;                       // wave-uniform exit; EXEC all-1s

  // A operand: lane<16 -> row m0+lane, K {0..15};
  //            lane>=16 -> row m0+lane-16, K {16..31}
  const int koff = (lane >> 4) << 4;         // 0 or 16
  const float* arow = A + (size_t)(m0 + (lane & 15)) * lda + koff;

  v8f acc[NTILES] = {};
  for (int kt = 0; kt < ktiles; ++kt) {
    const int kb = kt << 5;
    // --- fetch all B operands for this k-step (independent b128 loads) ---
    ABvec bv[NTILES];
#pragma unroll
    for (int nt = 0; nt < NTILES; ++nt) {
      const uint4* bp = (const uint4*)(Bp + (((size_t)nt * ktiles + kt) * 32 + lane) * 8);
      uint4 b0 = bp[0], b1 = bp[1];
      bv[nt].u[0] = b0.x; bv[nt].u[1] = b0.y; bv[nt].u[2] = b0.z; bv[nt].u[3] = b0.w;
      bv[nt].u[4] = b1.x; bv[nt].u[5] = b1.y; bv[nt].u[6] = b1.z; bv[nt].u[7] = b1.w;
    }
    // --- load + convert A operand ---
    float4 a0 = *(const float4*)(arow + kb);
    float4 a1 = *(const float4*)(arow + kb + 4);
    float4 a2 = *(const float4*)(arow + kb + 8);
    float4 a3 = *(const float4*)(arow + kb + 12);
    ABvec av;
    av.u[0] = f2bf(a0.x) | (f2bf(a0.y) << 16);
    av.u[1] = f2bf(a0.z) | (f2bf(a0.w) << 16);
    av.u[2] = f2bf(a1.x) | (f2bf(a1.y) << 16);
    av.u[3] = f2bf(a1.z) | (f2bf(a1.w) << 16);
    av.u[4] = f2bf(a2.x) | (f2bf(a2.y) << 16);
    av.u[5] = f2bf(a2.z) | (f2bf(a2.w) << 16);
    av.u[6] = f2bf(a3.x) | (f2bf(a3.y) << 16);
    av.u[7] = f2bf(a3.z) | (f2bf(a3.w) << 16);
    // --- WMMA burst: 4 independent accumulators, shared A ---
#pragma unroll
    for (int nt = 0; nt < NTILES; ++nt) {
      acc[nt] = __builtin_amdgcn_wmma_f32_16x16x32_bf16(
          false, av.v, false, bv[nt].v, (short)0, acc[nt], false, false);
    }
  }
  // C layout: lane<16 -> (M=v, N=lane); lane>=16 -> (M=v+8, N=lane-16)
  const int col   = lane & 15;
  const int rbase = m0 + ((lane >> 4) << 3);
#pragma unroll
  for (int nt = 0; nt < NTILES; ++nt)
#pragma unroll
    for (int v = 0; v < 8; ++v)
      C[(size_t)(rbase + v) * HPAD + nt * 16 + col] = acc[nt][v];
}

// ---------------------------------------------------------------------------
// Degree / normalization
// ---------------------------------------------------------------------------
__global__ void deg_init_kernel(float* deg, int Nn) {
  int i = blockIdx.x * blockDim.x + threadIdx.x;
  if (i < Nn) deg[i] = 1.0f;                 // self loop
}
__global__ void deg_acc_kernel(const int* __restrict__ ei, int E, float* deg) {
  int e = blockIdx.x * blockDim.x + threadIdx.x;
  if (e < E) unsafeAtomicAdd(&deg[ei[E + e]], 1.0f);   // dst = ei[1][e]
}
__global__ void deg_rsqrt_kernel(float* deg, int Nn) {
  int i = blockIdx.x * blockDim.x + threadIdx.x;
  if (i < Nn) deg[i] = rsqrtf(deg[i]);
}

// ---------------------------------------------------------------------------
// out[i][f] = H[i][f]*dinv[i]^2 for f<F else 0  (self loop fused with zeroing)
// ---------------------------------------------------------------------------
__global__ void init_agg_kernel(const float* __restrict__ H,
                                float* __restrict__ out,
                                const float* __restrict__ dinv,
                                int F, int Nn) {
  size_t idx = (size_t)blockIdx.x * blockDim.x + threadIdx.x;
  if (idx >= (size_t)Nn * HPAD) return;
  int i = (int)(idx >> 6);
  int f = (int)(idx & 63);
  float d = dinv[i];
  out[idx] = (f < F) ? H[idx] * d * d : 0.0f;
}

// ---------------------------------------------------------------------------
// Edge aggregation: one wave32 per edge; lanes cover features f=lane, lane+32.
// H (25.6 MB) is L2-resident on MI455X (192 MB L2), so gathers and the f32
// atomic adds resolve at L2 bandwidth.
// ---------------------------------------------------------------------------
__global__ void __launch_bounds__(256)
edge_agg_kernel(const int* __restrict__ ei, int E,
                const float* __restrict__ H, float* __restrict__ out,
                const float* __restrict__ dinv, int F) {
  int e = blockIdx.x * (blockDim.x >> 5) + (threadIdx.x >> 5);
  if (e >= E) return;
  int lane = threadIdx.x & 31;
  int src = __builtin_amdgcn_readfirstlane(ei[e]);
  int dst = __builtin_amdgcn_readfirstlane(ei[E + e]);
  float w = dinv[src] * dinv[dst];
  const float* hs = H + (size_t)src * HPAD;
  float* od = out + (size_t)dst * HPAD;
  unsafeAtomicAdd(&od[lane], hs[lane] * w);              // F >= 32 always
  if (lane < F - 32) unsafeAtomicAdd(&od[lane + 32], hs[lane + 32] * w);
}

// ---------------------------------------------------------------------------
// buf[i][f] = relu(buf[i][f] + b[f]) for f<F, 0 otherwise (keeps K-padding 0)
// ---------------------------------------------------------------------------
__global__ void bias_relu_kernel(float* buf, const float* __restrict__ b,
                                 int F, int Nn) {
  size_t idx = (size_t)blockIdx.x * blockDim.x + threadIdx.x;
  if (idx >= (size_t)Nn * HPAD) return;
  int f = (int)(idx & 63);
  buf[idx] = (f < F) ? fmaxf(buf[idx] + b[f], 0.0f) : 0.0f;
}

// ---------------------------------------------------------------------------
// Row log-softmax over C=40 classes; one wave32 per node.
// ---------------------------------------------------------------------------
__global__ void __launch_bounds__(256)
logsoftmax_kernel(const float* __restrict__ agg, const float* __restrict__ b2,
                  float* __restrict__ out, int Nn) {
  int i = blockIdx.x * (blockDim.x >> 5) + (threadIdx.x >> 5);
  if (i >= Nn) return;
  int lane = threadIdx.x & 31;
  const float* r = agg + (size_t)i * HPAD;
  float a = r[lane] + b2[lane];                           // f = lane (<40)
  float b = (lane < F2 - 32) ? r[lane + 32] + b2[lane + 32]
                             : -__builtin_inff();         // f = lane+32
  float m = fmaxf(a, b);
#pragma unroll
  for (int off = 16; off; off >>= 1) m = fmaxf(m, __shfl_xor(m, off, 32));
  float s = __expf(a - m) + ((lane < F2 - 32) ? __expf(b - m) : 0.0f);
#pragma unroll
  for (int off = 16; off; off >>= 1) s += __shfl_xor(s, off, 32);
  float ls = __logf(s);
  float* o = out + (size_t)i * F2;
  o[lane] = a - m - ls;
  if (lane < F2 - 32) o[lane + 32] = b - m - ls;
}

// ---------------------------------------------------------------------------
// Host launcher
// ---------------------------------------------------------------------------
static inline size_t align256(size_t x) { return (x + 255) & ~(size_t)255; }

extern "C" void kernel_launch(void* const* d_in, const int* in_sizes, int n_in,
                              void* d_out, int out_size, void* d_ws, size_t ws_size,
                              hipStream_t stream) {
  const float* x  = (const float*)d_in[0];   // [N, 512]
  const float* W1 = (const float*)d_in[1];   // [512, 50]
  const float* b1 = (const float*)d_in[2];   // [50]
  const float* W2 = (const float*)d_in[3];   // [50, 40]
  const float* b2 = (const float*)d_in[4];   // [40]
  const int*   ei = (const int*)d_in[5];     // [2, E]
  float* out = (float*)d_out;

  const int N = in_sizes[0] / FIN;
  const int E = in_sizes[5] / 2;
  const int kt1 = FIN / 32;                  // 16
  const int kt2 = HPAD / 32;                 // 2

  char* ws = (char*)d_ws;
  size_t off = 0;
  float* dinv = (float*)(ws + off);          off = align256(off + (size_t)N * 4);
  unsigned int* W1p = (unsigned int*)(ws + off);
  off = align256(off + (size_t)NTILES * kt1 * 32 * 8 * 4);
  unsigned int* W2p = (unsigned int*)(ws + off);
  off = align256(off + (size_t)NTILES * kt2 * 32 * 8 * 4);
  float* bufA = (float*)(ws + off);          off = align256(off + (size_t)N * HPAD * 4);
  float* bufB = (float*)(ws + off);          off = align256(off + (size_t)N * HPAD * 4);

  const int BT = 256;
  const int elemBlocks = (int)(((size_t)N * HPAD + BT - 1) / BT);
  const int mWaves = (N + 15) / 16;
  const int gemmBlocks = (mWaves + 7) / 8;
  const int edgeBlocks = (E + 7) / 8;
  const int nodeWaveBlocks = (N + 7) / 8;

  // 1) normalization coefficients
  deg_init_kernel<<<(N + BT - 1) / BT, BT, 0, stream>>>(dinv, N);
  deg_acc_kernel<<<(E + BT - 1) / BT, BT, 0, stream>>>(ei, E, dinv);
  deg_rsqrt_kernel<<<(N + BT - 1) / BT, BT, 0, stream>>>(dinv, N);

  // 2) pack weights into WMMA B-operand layout (bf16)
  pack_w_kernel<<<(NTILES * kt1 * 32 + BT - 1) / BT, BT, 0, stream>>>(W1, FIN, F1, kt1, W1p);
  pack_w_kernel<<<(NTILES * kt2 * 32 + BT - 1) / BT, BT, 0, stream>>>(W2, F1, F2, kt2, W2p);

  // 3) layer 1: h1 = x @ W1  (WMMA)
  wmma_gemm_kernel<<<gemmBlocks, BT, 0, stream>>>(x, FIN, kt1, W1p, bufA, N);

  // 4) aggregation: self loops + edges, then bias+ReLU
  init_agg_kernel<<<elemBlocks, BT, 0, stream>>>(bufA, bufB, dinv, F1, N);
  edge_agg_kernel<<<edgeBlocks, BT, 0, stream>>>(ei, E, bufA, bufB, dinv, F1);
  bias_relu_kernel<<<elemBlocks, BT, 0, stream>>>(bufB, b1, F1, N);

  // 5) layer 2: h2 = relu_agg1 @ W2  (WMMA, K padded 50->64 with zeros)
  wmma_gemm_kernel<<<gemmBlocks, BT, 0, stream>>>(bufB, HPAD, kt2, W2p, bufA, N);

  // 6) aggregation for layer 2
  init_agg_kernel<<<elemBlocks, BT, 0, stream>>>(bufA, bufB, dinv, F2, N);
  edge_agg_kernel<<<edgeBlocks, BT, 0, stream>>>(ei, E, bufA, bufB, dinv, F2);

  // 7) + b2, row log-softmax -> d_out [N, 40]
  logsoftmax_kernel<<<nodeWaveBlocks, BT, 0, stream>>>(bufB, b2, out, N);
}
